// OctreeAttention_20615843021051
// MI455X (gfx1250) — matrix operand order repaired and checked by reading, not compiled, not run on hardware
//
#include <hip/hip_runtime.h>

#define KP   24
#define DIL  4
#define CD   256
#define NQ   (3 * CD)        // 768
#define NH   8
#define HD   32
#define NT   196608
#define NW   (NT / KP)       // 8192 windows
#define PB   38
#define RN   77              // 2*PB+1

typedef __attribute__((ext_vector_type(16))) __bf16 v16bf;
typedef __attribute__((ext_vector_type(8)))  __bf16 v8bf;
typedef __attribute__((ext_vector_type(8)))  float  v8f;
typedef __attribute__((ext_vector_type(4)))  int    v4i;

#if defined(__HIP_DEVICE_COMPILE__) && __has_builtin(__builtin_amdgcn_global_load_async_to_lds_b128)
#define ASYNC_LDS 1
#else
#define ASYNC_LDS 0
#endif

#if ASYNC_LDS
typedef __attribute__((address_space(1))) v4i* gv4i_p;
typedef __attribute__((address_space(3))) v4i* lv4i_p;
#endif

__device__ __forceinline__ void wait_async0() {
#if defined(__HIP_DEVICE_COMPILE__) && __has_builtin(__builtin_amdgcn_s_wait_asynccnt)
  __builtin_amdgcn_s_wait_asynccnt(0);
#elif defined(__HIP_DEVICE_COMPILE__)
  asm volatile("s_wait_asynccnt 0x0" ::: "memory");
#endif
}

__device__ __forceinline__ v8f wmma_bf16(v16bf a, v16bf b, v8f c) {
  return __builtin_amdgcn_wmma_f32_16x16x32_bf16(false, a, false, b, (short)0, c,
                                                 false, false);
}

// A operand (16x32 bf16, row-major source): lane half 0 holds K 0-7 & 16-23,
// half 1 holds K 8-15 & 24-31 -> two 16B loads.
__device__ __forceinline__ v16bf load_afrag(const __bf16* rowp, int hf) {
  union { v16bf v; v8bf h[2]; } u;
  u.h[0] = *(const v8bf*)(rowp + hf * 8);
  u.h[1] = *(const v8bf*)(rowp + 16 + hf * 8);
  return u.v;
}
// B operand (32x16 bf16, stored transposed [N][K]): lane half 0 holds
// K 0-7 & 16-23, half 1 holds K 8-15 & 24-31 of the lane's column.
__device__ __forceinline__ v16bf load_bfrag(const __bf16* colp, int hf) {
  union { v16bf v; v8bf h[2]; } u;
  u.h[0] = *(const v8bf*)(colp + hf * 8);
  u.h[1] = *(const v8bf*)(colp + 16 + hf * 8);
  return u.v;
}

// ---------------- Kernel 1: fused dilation-shuffle + QKV GEMM --------------
// 128x128 block tile, BK=32, double-buffered LDS, wave tile 32x64 (8 WMMA/step)
__global__ __launch_bounds__(256) void oa_qkv(const float* __restrict__ data,
                                              const float* __restrict__ wqkv,
                                              const float* __restrict__ bqkv,
                                              __bf16* __restrict__ qb,
                                              __bf16* __restrict__ kb,
                                              __bf16* __restrict__ vb) {
  __shared__ __align__(32) __bf16 As[2][128][32];
  __shared__ __align__(32) __bf16 Bs[2][128][32];   // transposed [n][k]
  const int tid  = threadIdx.x;
  const int lane = tid & 31, wv = tid >> 5;
  const int wm = wv & 3, wn = wv >> 2;              // wave: rows wm*32, cols wn*64
  const int row0 = blockIdx.x * 128;
  const int col0 = blockIdx.y * 128;
  const int r16 = lane & 15, hf = lane >> 4;

  // A loader: hoisted permuted-row offsets (dilation shuffle)
  const int ac  = (tid & 7) * 4;
  const int arb = tid >> 3;                          // 0..31
  size_t goff[4];
#pragma unroll
  for (int i = 0; i < 4; ++i) {
    const int xr = row0 + arb + i * 32;
    const int w = xr / KP, t = xr - w * KP;
    goff[i] = (size_t)((w >> 2) * (KP * DIL) + t * DIL + (w & 3)) * CD + ac;
  }
  // B loader indices
  const int bn = (tid & 31) * 4;
  const int bk = tid >> 5;                           // 0..7

  float4 ra[4], rb4[4];
  auto ldG = [&](int k0) {
#pragma unroll
    for (int i = 0; i < 4; ++i) ra[i] = *(const float4*)(data + goff[i] + k0);
#pragma unroll
    for (int i = 0; i < 4; ++i)
      rb4[i] = *(const float4*)(wqkv + (size_t)(k0 + bk + i * 8) * NQ + col0 + bn);
  };
  auto stL = [&](int b) {
#pragma unroll
    for (int i = 0; i < 4; ++i) {
      const int r = arb + i * 32;
      As[b][r][ac + 0] = (__bf16)ra[i].x;
      As[b][r][ac + 1] = (__bf16)ra[i].y;
      As[b][r][ac + 2] = (__bf16)ra[i].z;
      As[b][r][ac + 3] = (__bf16)ra[i].w;
    }
#pragma unroll
    for (int i = 0; i < 4; ++i) {
      const int k = bk + i * 8;
      Bs[b][bn + 0][k] = (__bf16)rb4[i].x;
      Bs[b][bn + 1][k] = (__bf16)rb4[i].y;
      Bs[b][bn + 2][k] = (__bf16)rb4[i].z;
      Bs[b][bn + 3][k] = (__bf16)rb4[i].w;
    }
  };

  v8f acc[2][4] = {};

  ldG(0);
  stL(0);
  __syncthreads();
#pragma unroll
  for (int k0 = 0; k0 < CD; k0 += 32) {
    const int buf = (k0 >> 5) & 1, nxt = buf ^ 1;
    const bool more = (k0 + 32) < CD;
    if (more) ldG(k0 + 32);          // global loads in flight over the WMMAs
    v16bf af[2], bfr[4];
#pragma unroll
    for (int mt = 0; mt < 2; ++mt)
      af[mt] = load_afrag(&As[buf][wm * 32 + mt * 16 + r16][0], hf);
#pragma unroll
    for (int nt = 0; nt < 4; ++nt)
      bfr[nt] = load_bfrag(&Bs[buf][wn * 64 + nt * 16 + r16][0], hf);
#pragma unroll
    for (int mt = 0; mt < 2; ++mt)
#pragma unroll
      for (int nt = 0; nt < 4; ++nt)
        acc[mt][nt] = wmma_bf16(af[mt], bfr[nt], acc[mt][nt]);
    if (more) stL(nxt);
    __syncthreads();
  }

  // epilogue: scalarized q/k/v routing, cheap w/t recurrence
  const float scale = 0.17677669529663689f;   // 32^-0.5
#pragma unroll
  for (int nt = 0; nt < 4; ++nt) {
    const int c  = col0 + wn * 64 + nt * 16 + r16;
    const int cb = __builtin_amdgcn_readfirstlane(c & ~15);  // wave-uniform
    const int part = cb >> 8;
    const int h    = (cb & 255) >> 5;
    const int e    = c & 31;
    const float bias = bqkv[c];
    __bf16* dst; float mul; int trp;
    if (part == 0)      { dst = qb; mul = scale; trp = 0; }
    else if (part == 1) { dst = kb; mul = 1.f;   trp = 0; }
    else                { dst = vb; mul = 1.f;   trp = 1; }
#pragma unroll
    for (int mt = 0; mt < 2; ++mt) {
      const int rb = row0 + wm * 32 + mt * 16 + hf * 8;
      const int w0 = rb / KP;
      const int t0 = rb - w0 * KP;
#pragma unroll
      for (int j = 0; j < 8; ++j) {
        const int cr = (t0 + j >= KP) ? 1 : 0;
        const int w = w0 + cr, t = t0 + j - KP * cr;
        const size_t base = (size_t)(w * NH + h) * 1024;
        const float v = acc[mt][nt][j] + bias;
        const int idx = trp ? (e * 32 + t) : (t * 32 + e);
        dst[base + idx] = (__bf16)(v * mul);
      }
    }
  }
}

// ---------------- Kernel 2: windowed attention, 1 wave = (window, head) ----
__global__ __launch_bounds__(256) void oa_attn(const __bf16* __restrict__ qb,
                                               const __bf16* __restrict__ kb,
                                               const __bf16* __restrict__ vb,
                                               const float* __restrict__ rtab,
                                               const int* __restrict__ rpos,
                                               const float* __restrict__ mask,
                                               __bf16* __restrict__ ab) {
  __shared__ float tab[3 * RN * NH];                 // RPE table, 231x8
  __shared__ float S[NH][KP][25];                    // padded vs bank conflicts
  __shared__ __align__(32) __bf16 P[NH][32][32];     // zero-padded probs
  const int tid  = threadIdx.x;
  const int lane = tid & 31, h = tid >> 5;
  const int w = blockIdx.x;
  const int r16 = lane & 15, hf = lane >> 4;

  for (int i = tid; i < 3 * RN * NH; i += 256) tab[i] = rtab[i];
  __syncthreads();

  const size_t wh = (size_t)(w * NH + h) * (32 * 32);

  // S = q * k^T   (hd=32 contraction, single WMMA step, 2x2 tiles)
  v8f sa[2][2] = {};
  {
    v16bf qf[2], kf[2];
#pragma unroll
    for (int mt = 0; mt < 2; ++mt)
      qf[mt] = load_afrag(qb + wh + (size_t)(mt * 16 + r16) * 32, hf);
#pragma unroll
    for (int nt = 0; nt < 2; ++nt)
      kf[nt] = load_bfrag(kb + wh + (size_t)(nt * 16 + r16) * 32, hf);
#pragma unroll
    for (int mt = 0; mt < 2; ++mt)
#pragma unroll
      for (int nt = 0; nt < 2; ++nt)
        sa[mt][nt] = wmma_bf16(qf[mt], kf[nt], sa[mt][nt]);
  }
#pragma unroll
  for (int mt = 0; mt < 2; ++mt)
#pragma unroll
    for (int nt = 0; nt < 2; ++nt) {
      const int cc = nt * 16 + r16;
      if (cc < KP) {
#pragma unroll
        for (int j = 0; j < 8; ++j) {
          const int rr = mt * 16 + hf * 8 + j;
          if (rr < KP) S[h][rr][cc] = sa[mt][nt][j];
        }
      }
    }
  __syncthreads();

  // per-row softmax with RPE bias + mask (lane = token row)
  if (lane < KP) {
    const int t = lane;
    const int*  rp = rpos + (size_t)(w * KP + t) * KP * 3;
    const float* mk = mask + (size_t)(w * KP + t) * KP;
    float sv[KP];
    float m = -3.0e38f;
#pragma unroll
    for (int j = 0; j < KP; ++j) {
      int a0 = rp[3 * j + 0], a1 = rp[3 * j + 1], a2 = rp[3 * j + 2];
      a0 = min(max(a0, -PB), PB);
      a1 = min(max(a1, -PB), PB);
      a2 = min(max(a2, -PB), PB);
      const float bias = tab[(a0 + PB) * NH + h]
                       + tab[(a1 + PB + RN) * NH + h]
                       + tab[(a2 + PB + 2 * RN) * NH + h];
      const float s = S[h][t][j] + bias + mk[j];
      sv[j] = s;
      m = fmaxf(m, s);
    }
    float sum = 0.f;
#pragma unroll
    for (int j = 0; j < KP; ++j) { const float e = __expf(sv[j] - m); sv[j] = e; sum += e; }
    const float inv = 1.f / sum;
#pragma unroll
    for (int j = 0; j < KP; ++j) P[h][t][j] = (__bf16)(sv[j] * inv);
#pragma unroll
    for (int j = KP; j < 32; ++j) P[h][t][j] = (__bf16)0.f;   // pad cols
  } else {
#pragma unroll
    for (int j = 0; j < 32; ++j) P[h][lane][j] = (__bf16)0.f; // pad rows
  }
  __syncthreads();

  // O = P * v
  v8f oacc[2][2] = {};
  {
    v16bf pf[2], vf[2];
#pragma unroll
    for (int mt = 0; mt < 2; ++mt)
      pf[mt] = load_afrag(&P[h][mt * 16 + r16][0], hf);
#pragma unroll
    for (int nt = 0; nt < 2; ++nt)
      vf[nt] = load_bfrag(vb + wh + (size_t)(nt * 16 + r16) * 32, hf);
#pragma unroll
    for (int mt = 0; mt < 2; ++mt)
#pragma unroll
      for (int nt = 0; nt < 2; ++nt)
        oacc[mt][nt] = wmma_bf16(pf[mt], vf[nt], oacc[mt][nt]);
  }
  // store with inverse dilation shuffle fused
  const int g0 = (w >> 2) * (KP * DIL) + (w & 3);
#pragma unroll
  for (int mt = 0; mt < 2; ++mt)
#pragma unroll
    for (int nt = 0; nt < 2; ++nt) {
      const int e = nt * 16 + r16;
#pragma unroll
      for (int j = 0; j < 8; ++j) {
        const int t = mt * 16 + hf * 8 + j;
        if (t < KP)
          ab[(size_t)(g0 + t * DIL) * CD + h * HD + e] = (__bf16)oacc[mt][nt][j];
      }
    }
}

// ---------------- Kernel 3: output projection GEMM -------------------------
// 128x128 block tile, BK=32, double-buffered; A tile copied with async-to-LDS
__global__ __launch_bounds__(256) void oa_proj(const __bf16* __restrict__ ab,
                                               const float* __restrict__ wproj,
                                               const float* __restrict__ bproj,
                                               float* __restrict__ out) {
  __shared__ __align__(32) __bf16 As[2][128][32];
  __shared__ __align__(32) __bf16 Bs[2][128][32];
  const int tid  = threadIdx.x;
  const int lane = tid & 31, wv = tid >> 5;
  const int wm = wv & 3, wn = wv >> 2;
  const int row0 = blockIdx.x * 128;
  const int col0 = blockIdx.y * 128;
  const int r16 = lane & 15, hf = lane >> 4;

  // A copy: 512 b128 chunks, 2 per thread (rows tid>>2 and +64)
  const int arow = tid >> 2;
  const int acol = (tid & 3) * 8;
  const __bf16* agp = ab + (size_t)(row0 + arow) * CD + acol;
  // B loader
  const int bn = (tid & 31) * 4;
  const int bk = tid >> 5;

  float4 rb4[4];
#if !ASYNC_LDS
  uint4 rau[2];
#endif
  auto ldG = [&](int k0, int b) {
#if ASYNC_LDS
#pragma unroll
    for (int i = 0; i < 2; ++i) {
      __builtin_amdgcn_global_load_async_to_lds_b128(
          (gv4i_p)(agp + (size_t)(i * 64) * CD + k0),
          (lv4i_p)&As[b][arow + i * 64][acol], 0, 0);
    }
#else
#pragma unroll
    for (int i = 0; i < 2; ++i)
      rau[i] = *(const uint4*)(agp + (size_t)(i * 64) * CD + k0);
#endif
#pragma unroll
    for (int i = 0; i < 4; ++i)
      rb4[i] = *(const float4*)(wproj + (size_t)(k0 + bk + i * 8) * CD + col0 + bn);
  };
  auto stL = [&](int b) {
#if !ASYNC_LDS
#pragma unroll
    for (int i = 0; i < 2; ++i) *(uint4*)&As[b][arow + i * 64][acol] = rau[i];
#endif
#pragma unroll
    for (int i = 0; i < 4; ++i) {
      const int k = bk + i * 8;
      Bs[b][bn + 0][k] = (__bf16)rb4[i].x;
      Bs[b][bn + 1][k] = (__bf16)rb4[i].y;
      Bs[b][bn + 2][k] = (__bf16)rb4[i].z;
      Bs[b][bn + 3][k] = (__bf16)rb4[i].w;
    }
#if ASYNC_LDS
    wait_async0();
#endif
  };

  v8f acc[2][4] = {};

  ldG(0, 0);
  stL(0);
  __syncthreads();
#pragma unroll
  for (int k0 = 0; k0 < CD; k0 += 32) {
    const int buf = (k0 >> 5) & 1, nxt = buf ^ 1;
    const bool more = (k0 + 32) < CD;
    if (more) ldG(k0 + 32, nxt);     // async A -> LDS[nxt] + B global loads
    v16bf af[2], bfr[4];
#pragma unroll
    for (int mt = 0; mt < 2; ++mt)
      af[mt] = load_afrag(&As[buf][wm * 32 + mt * 16 + r16][0], hf);
#pragma unroll
    for (int nt = 0; nt < 4; ++nt)
      bfr[nt] = load_bfrag(&Bs[buf][wn * 64 + nt * 16 + r16][0], hf);
#pragma unroll
    for (int mt = 0; mt < 2; ++mt)
#pragma unroll
      for (int nt = 0; nt < 4; ++nt)
        acc[mt][nt] = wmma_bf16(af[mt], bfr[nt], acc[mt][nt]);
    if (more) stL(nxt);
    __syncthreads();
  }

#pragma unroll
  for (int nt = 0; nt < 4; ++nt) {
    const int c = col0 + wn * 64 + nt * 16 + r16;
    const float bias = bproj[c];
#pragma unroll
    for (int mt = 0; mt < 2; ++mt) {
      const int rb = row0 + wm * 32 + mt * 16 + hf * 8;
#pragma unroll
      for (int j = 0; j < 8; ++j)
        out[(size_t)(rb + j) * CD + c] = acc[mt][nt][j] + bias;
    }
  }
}

extern "C" void kernel_launch(void* const* d_in, const int* in_sizes, int n_in,
                              void* d_out, int out_size, void* d_ws, size_t ws_size,
                              hipStream_t stream) {
  const float* data  = (const float*)d_in[0];
  const float* mask  = (const float*)d_in[1];
  const float* wqkv  = (const float*)d_in[2];
  const float* bqkv  = (const float*)d_in[3];
  const float* wproj = (const float*)d_in[4];
  const float* bproj = (const float*)d_in[5];
  const float* rtab  = (const float*)d_in[6];
  const int*   rpos  = (const int*)d_in[7];
  float* out = (float*)d_out;

  char* ws = (char*)d_ws;
  const size_t padbuf = (size_t)NW * NH * 32 * 32 * 2;   // 134,217,728 B each
  __bf16* qb = (__bf16*)(ws);
  __bf16* kb = (__bf16*)(ws + padbuf);
  __bf16* vb = (__bf16*)(ws + 2 * padbuf);
  __bf16* ab = (__bf16*)(ws + 3 * padbuf);               // NT*CD bf16

  oa_qkv <<<dim3(NT / 128, NQ / 128), 256, 0, stream>>>(data, wqkv, bqkv, qb, kb, vb);
  oa_attn<<<dim3(NW), 256, 0, stream>>>(qb, kb, vb, rtab, rpos, mask, ab);
  oa_proj<<<dim3(NT / 128, CD / 128), 256, 0, stream>>>(ab, wproj, bproj, out);
}
